// CustomRNN_31344671326418
// MI455X (gfx1250) — matrix operand order repaired and checked by reading, not compiled, not run on hardware
//
#include <hip/hip_runtime.h>
#include <math.h>

// ---------------------------------------------------------------------------
// CustomRNN on MI455X (gfx1250, wave32, WMMA bf16 16x16x32, fp32 accum)
//   h_{t+1} = tanh( x[:,t,:] @ Wxh + h_t @ Whh + b_xh + b_hh )   (fused K=1536)
//   out     = h_T @ Why + b_y
// Weights pre-transposed to bf16 [N][K] so A and B tiles stage identically:
// contiguous global_load_b128 -> contiguous ds_store_b128 (no scatter).
// ---------------------------------------------------------------------------

#define B_    1024
#define T_    128
#define IN_   512
#define HID_  1024
#define NC_   10

#define BM    64        // block tile M (batch rows)
#define BN    128       // block tile N (hidden cols)
#define BK    32        // K chunk (one bf16 WMMA K-depth)
#define KS    40        // padded LDS row stride in bf16 elems (80B: 16B-aligned, rotates banks)

typedef __attribute__((ext_vector_type(16))) __bf16 v16bf;
typedef __attribute__((ext_vector_type(8)))  float  v8f;

union FragAB { v16bf v; uint4 q[2]; };
union FragC  { v8f   v; float f[8]; };

__device__ __forceinline__ unsigned short f2bf(float f) {
    unsigned u = __float_as_uint(f);
    u += 0x7FFFu + ((u >> 16) & 1u);          // round-to-nearest-even
    return (unsigned short)(u >> 16);
}
__device__ __forceinline__ float bf2f(unsigned short h) {
    return __uint_as_float(((unsigned)h) << 16);
}

// ---------------------------------------------------------------------------
// Init 1: LDS-tiled fp32 -> bf16 transpose:  WT[n*K + k] = bf16(W[k*N + n])
// grid (K/32, N/32), 256 threads (32x8), coalesced loads AND stores.
// ---------------------------------------------------------------------------
__global__ __launch_bounds__(256)
void transpose_bf16_kernel(const float* __restrict__ W, unsigned short* __restrict__ WT,
                           int K, int N)
{
    __shared__ unsigned short tile[32][33];    // +1 pad: conflict-free transpose
    const int k0 = blockIdx.x * 32, n0 = blockIdx.y * 32;
    const int tx = threadIdx.x & 31, ty = threadIdx.x >> 5;
    #pragma unroll
    for (int r = 0; r < 4; ++r)
        tile[ty + 8 * r][tx] = f2bf(W[(size_t)(k0 + ty + 8 * r) * N + n0 + tx]);
    __syncthreads();
    #pragma unroll
    for (int r = 0; r < 4; ++r)
        WT[(size_t)(n0 + ty + 8 * r) * K + k0 + tx] = tile[tx][ty + 8 * r];
}

// Init 2: zero h0 (workspace is poisoned 0xAA before timing). 512 blocks x 256.
__global__ __launch_bounds__(256)
void zero_h_kernel(uint4* __restrict__ h0)
{
    h0[(size_t)blockIdx.x * 256 + threadIdx.x] = uint4{0u, 0u, 0u, 0u};
}

// ---------------------------------------------------------------------------
// One recurrence step: fused GEMM over K = IN (x@WxhT phase) + HID (h@WhhT phase)
// 64x128 block tile, 8 waves (2x4), wave tile 32x32 -> 4 WMMAs per K chunk.
// Double-buffered LDS; global loads for chunk it+1 overlap WMMA on chunk it.
// ---------------------------------------------------------------------------
__global__ __launch_bounds__(256)
void rnn_step_kernel(const float* __restrict__ x, int tstep,
                     const unsigned short* __restrict__ WxhT,   // [HID][IN] bf16
                     const unsigned short* __restrict__ WhhT,   // [HID][HID] bf16
                     const float* __restrict__ b_xh,
                     const float* __restrict__ b_hh,
                     const unsigned short* __restrict__ h_in,   // [B][HID] bf16
                     unsigned short* __restrict__ h_out)        // [B][HID] bf16
{
    __shared__ unsigned short As[2][BM][KS];   // [M][K] bf16, row stride 80B
    __shared__ unsigned short Bs[2][BN][KS];   // [N][K] bf16 (weights already transposed)

    const int tid  = threadIdx.x;
    const int bn0  = blockIdx.x * BN;
    const int bm0  = blockIdx.y * BM;
    const int lane = tid & 31;
    const int wave = tid >> 5;
    const int lo   = lane & 15;
    const int hi   = lane >> 4;
    const int wm0  = (wave >> 2) * 32;         // wave M offset: 0 / 32
    const int wn0  = (wave & 3) * 32;          // wave N offset: 0/32/64/96

    const int P1  = IN_ / BK;                  // 16 iters of phase 1 (x @ Wxh)
    const int NIT = (IN_ + HID_) / BK;         // 48 total K iterations

    // Staging slots (all contiguous 8-bf16 groups -> one uint4 each):
    //   A: 64x32 = 2048 bf16 -> 1 group/thread
    //   B: 128x32 = 4096 bf16 -> 2 groups/thread
    const int amA = tid >> 2;                  // A: m in [0,64)
    const int akA = (tid & 3) * 8;             //    k in {0,8,16,24}
    const int nB0 = tid >> 2;                  // B group 0: n in [0,64)
    const int nB1 = (tid + 256) >> 2;          // B group 1: n in [64,128)
    const int kB  = (tid & 3) * 8;             //    k in {0,8,16,24}

    uint4 aReg, bReg0, bReg1;

    auto load_tile = [&](int it) {
        if (it < P1) {
            const int k0 = it * BK;
            // A source: x[bm0+m, tstep, k0+akA .. +7]  (fp32 -> bf16)
            const float* p = x + (size_t)(bm0 + amA) * (T_ * IN_)
                               + (size_t)tstep * IN_ + k0 + akA;
            float4 fa = *(const float4*)p;
            float4 fb = *(const float4*)(p + 4);
            aReg.x = (unsigned)f2bf(fa.x) | ((unsigned)f2bf(fa.y) << 16);
            aReg.y = (unsigned)f2bf(fa.z) | ((unsigned)f2bf(fa.w) << 16);
            aReg.z = (unsigned)f2bf(fb.x) | ((unsigned)f2bf(fb.y) << 16);
            aReg.w = (unsigned)f2bf(fb.z) | ((unsigned)f2bf(fb.w) << 16);
            if (k0 + BK < IN_) __builtin_prefetch(p + BK, 0, 3);  // near-scope prefetch
            bReg0 = *(const uint4*)(WxhT + (size_t)(bn0 + nB0) * IN_ + k0 + kB);
            bReg1 = *(const uint4*)(WxhT + (size_t)(bn0 + nB1) * IN_ + k0 + kB);
        } else {
            const int k0 = (it - P1) * BK;
            aReg  = *(const uint4*)(h_in + (size_t)(bm0 + amA) * HID_ + k0 + akA);
            bReg0 = *(const uint4*)(WhhT + (size_t)(bn0 + nB0) * HID_ + k0 + kB);
            bReg1 = *(const uint4*)(WhhT + (size_t)(bn0 + nB1) * HID_ + k0 + kB);
        }
    };

    auto store_tile = [&](int buf) {
        *(uint4*)&As[buf][amA][akA] = aReg;    // all ds_store_b128, no scatter
        *(uint4*)&Bs[buf][nB0][kB]  = bReg0;
        *(uint4*)&Bs[buf][nB1][kB]  = bReg1;
    };

    FragC acc[2][2];
    #pragma unroll
    for (int mt = 0; mt < 2; ++mt)
        #pragma unroll
        for (int nt = 0; nt < 2; ++nt)
            #pragma unroll
            for (int r = 0; r < 8; ++r) acc[mt][nt].f[r] = 0.0f;

    load_tile(0);
    store_tile(0);
    __syncthreads();

    for (int it = 0; it < NIT; ++it) {
        const int buf = it & 1;
        if (it + 1 < NIT) load_tile(it + 1);   // overlap global latency with WMMA

        // A fragment (16x32 bf16): lane(lo,hi) = row M=lo, K = {8h..8h+7, 16+8h..+7}
        FragAB af[2], bfr[2];
        #pragma unroll
        for (int mt = 0; mt < 2; ++mt) {
            const unsigned short* r = &As[buf][wm0 + mt * 16 + lo][0];
            af[mt].q[0] = *(const uint4*)(r + 8 * hi);          // ds_load_b128
            af[mt].q[1] = *(const uint4*)(r + 16 + 8 * hi);
        }
        // B fragment (32x16 bf16): lane(lo,hi) = col N=lo, K = {8h..8h+7, 16+8h..+7}
        #pragma unroll
        for (int nt = 0; nt < 2; ++nt) {
            const unsigned short* r = &Bs[buf][wn0 + nt * 16 + lo][0];
            bfr[nt].q[0] = *(const uint4*)(r + 8 * hi);
            bfr[nt].q[1] = *(const uint4*)(r + 16 + 8 * hi);
        }

        #pragma unroll
        for (int mt = 0; mt < 2; ++mt)
            #pragma unroll
            for (int nt = 0; nt < 2; ++nt)
                acc[mt][nt].v = __builtin_amdgcn_wmma_f32_16x16x32_bf16(
                    false, af[mt].v, false, bfr[nt].v, (short)0, acc[mt][nt].v,
                    false, false);

        if (it + 1 < NIT) {
            __syncthreads();
            store_tile(buf ^ 1);
            __syncthreads();
        }
    }

    // Epilogue: bias + tanh, write bf16 h for next step.
    // C/D layout: VGPR r holds M = r + 8*hi, N = lo.
    #pragma unroll
    for (int nt = 0; nt < 2; ++nt) {
        const int n = bn0 + wn0 + nt * 16 + lo;
        const float bias = b_xh[n] + b_hh[n];
        #pragma unroll
        for (int mt = 0; mt < 2; ++mt) {
            #pragma unroll
            for (int r = 0; r < 8; ++r) {
                const int m = bm0 + wm0 + mt * 16 + r + 8 * hi;
                h_out[(size_t)m * HID_ + n] = f2bf(tanhf(acc[mt][nt].f[r] + bias));
            }
        }
    }
}

// ---------------------------------------------------------------------------
// Head: out[b, c] = sum_j bf2f(hT[b,j]) * Why[j,c] + b_y[c]   (NC=10, skinny)
// ---------------------------------------------------------------------------
__global__ __launch_bounds__(256)
void rnn_head_kernel(const unsigned short* __restrict__ hT,
                     const float* __restrict__ Why,
                     const float* __restrict__ b_y,
                     float* __restrict__ out)
{
    const int row = blockIdx.x;
    float p[NC_];
    #pragma unroll
    for (int c = 0; c < NC_; ++c) p[c] = 0.0f;

    for (int j = threadIdx.x; j < HID_; j += 256) {
        const float hv = bf2f(hT[(size_t)row * HID_ + j]);
        #pragma unroll
        for (int c = 0; c < NC_; ++c) p[c] += hv * Why[j * NC_ + c];
    }
    // wave32 shuffle reduction
    #pragma unroll
    for (int c = 0; c < NC_; ++c)
        for (int off = 16; off > 0; off >>= 1)
            p[c] += __shfl_down(p[c], off, 32);

    __shared__ float sm[8][NC_];
    const int wave = threadIdx.x >> 5, lane = threadIdx.x & 31;
    if (lane == 0)
        #pragma unroll
        for (int c = 0; c < NC_; ++c) sm[wave][c] = p[c];
    __syncthreads();

    if (threadIdx.x < NC_) {
        float s = b_y[threadIdx.x];
        #pragma unroll
        for (int w = 0; w < 8; ++w) s += sm[w][threadIdx.x];
        out[(size_t)row * NC_ + threadIdx.x] = s;
    }
}

// ---------------------------------------------------------------------------
extern "C" void kernel_launch(void* const* d_in, const int* in_sizes, int n_in,
                              void* d_out, int out_size, void* d_ws, size_t ws_size,
                              hipStream_t stream) {
    (void)in_sizes; (void)n_in; (void)out_size; (void)ws_size;
    const float* x   = (const float*)d_in[0];
    const float* Wxh = (const float*)d_in[1];
    const float* bxh = (const float*)d_in[2];
    const float* Whh = (const float*)d_in[3];
    const float* bhh = (const float*)d_in[4];
    const float* Why = (const float*)d_in[5];
    const float* by  = (const float*)d_in[6];
    float* out = (float*)d_out;

    unsigned char* ws = (unsigned char*)d_ws;
    unsigned short* WxhT = (unsigned short*)(ws);                         // 1 MB  [HID][IN]
    unsigned short* WhhT = (unsigned short*)(ws + (size_t)1 * (1u<<20));  // 2 MB  [HID][HID]
    unsigned short* hA   = (unsigned short*)(ws + (size_t)3 * (1u<<20));  // 2 MB  [B][HID]
    unsigned short* hB   = (unsigned short*)(ws + (size_t)5 * (1u<<20));  // 2 MB  [B][HID]

    transpose_bf16_kernel<<<dim3(IN_ / 32, HID_ / 32), dim3(256), 0, stream>>>(
        Wxh, WxhT, IN_, HID_);
    transpose_bf16_kernel<<<dim3(HID_ / 32, HID_ / 32), dim3(256), 0, stream>>>(
        Whh, WhhT, HID_, HID_);
    zero_h_kernel<<<dim3((B_ * HID_ * 2) / (16 * 256)), dim3(256), 0, stream>>>(
        (uint4*)hA);

    unsigned short* hin  = hA;
    unsigned short* hout = hB;
    for (int t = 0; t < T_; ++t) {
        rnn_step_kernel<<<dim3(HID_ / BN, B_ / BM), dim3(256), 0, stream>>>(
            x, t, WxhT, WhhT, bxh, bhh, hin, hout);
        unsigned short* tmp = hin; hin = hout; hout = tmp;
    }

    rnn_head_kernel<<<dim3(B_), dim3(256), 0, stream>>>(hin, Why, by, out);
}
